// EquivariantProductBasisWithSelfMagmomBlock_42125039239738
// MI455X (gfx1250) — compile-verified, matrix operand order checked by reading, not run on hardware
//
#include <hip/hip_runtime.h>
#include <hip/hip_bf16.h>

typedef __attribute__((ext_vector_type(16))) __bf16 v16bf;
typedef __attribute__((ext_vector_type(8)))  __bf16 v8bf;
typedef __attribute__((ext_vector_type(2)))  __bf16 v2bf;
typedef __attribute__((ext_vector_type(8)))  float  v8f;

#define WMMA_BF16(A, B, C) __builtin_amdgcn_wmma_f32_16x16x32_bf16(false, (A), false, (B), (short)0, (C), false, false)

static constexpr int N_NODES = 65536;
static constexpr int C = 128;
static constexpr int WAVES = 4;

// Packed-B fragment offsets (in bf16 elements). Each fragment = 32 lanes * 16 elems = 512.
static constexpr int OFF_BM1 = 0;                      // w_mlp1: 1 kstep * 4 jt
static constexpr int OFF_BM2 = OFF_BM1 + 4 * 512;      // w_mlp2: 2 * 4
static constexpr int OFF_BM3 = OFF_BM2 + 8 * 512;      // w_mlp3: 2 * 4
static constexpr int OFF_BM4 = OFF_BM3 + 8 * 512;      // w_mlp4: 2 * 32
static constexpr int OFF_BG0 = OFF_BM4 + 64 * 512;     // [W_l0;Wo0]: 12 * 8
static constexpr int OFF_BG1 = OFF_BG0 + 96 * 512;     // [W_l1;Wo1]: 12 * 8

// LDS carve (bf16 elements)
static constexpr int TPW_ELEMS = WAVES * 16 * 512;          // 32768
static constexpr int A_ELEMS   = WAVES * 4 * 16 * 384;      // 98304
static constexpr int H_ELEMS   = WAVES * 16 * 64;           // 4096
static constexpr size_t LDS_BYTES = (size_t)(TPW_ELEMS + A_ELEMS + H_ELEMS) * 2;  // 270336

static __device__ __forceinline__ v16bf cat8(v8bf lo, v8bf hi) {
    return __builtin_shufflevector(lo, hi, 0, 1, 2, 3, 4, 5, 6, 7, 8, 9, 10, 11, 12, 13, 14, 15);
}

// SiLU with v_rcp_f32 instead of the precise-divide expansion.
static __device__ __forceinline__ float silu(float x) {
    return x * __builtin_amdgcn_rcpf(1.0f + __expf(-x));
}

// Pack a (rows0+rows1) x (njt*16) fp32 matrix (two stacked sources, zero-padded
// K) into bf16 WMMA B fragments: frag f=(kstep*njt+j), per lane 16 contiguous
// bf16 laid out in the 16-bit-B VGPR order (khalf/pair interleave).
__global__ void pack_b_kernel(const float* __restrict__ src0, int rows0,
                              const float* __restrict__ src1, int rows1,
                              int njt, __bf16* __restrict__ dst) {
    int f = blockIdx.x;
    int kstep = f / njt;
    int j = f % njt;
    int lane = threadIdx.x & 31;
    int ncols = njt * 16;
    int col = j * 16 + (lane & 15);
    int kh = (lane >> 4) * 8;
#pragma unroll
    for (int e = 0; e < 16; ++e) {
        int klocal = (e < 8) ? (kh + e) : (kh + 16 + (e - 8));
        int k = kstep * 32 + klocal;
        float val = 0.0f;
        if (k < rows0) val = src0[(size_t)k * ncols + col];
        else if (k < rows0 + rows1) val = src1[(size_t)(k - rows0) * ncols + col];
        dst[(size_t)f * 512 + lane * 16 + e] = (__bf16)val;
    }
}

__global__ __launch_bounds__(128) void fused_kernel(
    const float* __restrict__ node_feats,    // N x C x 4
    const float* __restrict__ sc,            // N x 4C
    const float* __restrict__ node_attrs,    // N x 10 (one-hot)
    const float* __restrict__ invf,          // N x 16
    const float* __restrict__ mga,           // N x 4
    const float* __restrict__ w_sc0,         // E x 5 x C
    const float* __restrict__ w_sc1,         // E x 4 x C
    const __bf16* __restrict__ bpack,        // packed B fragments
    float* __restrict__ out)                 // N x 4C
{
    extern __shared__ __bf16 lds[];
    __bf16* tpwL = lds;                       // [WAVES*16][512]
    __bf16* aL   = lds + TPW_ELEMS;           // [WAVES][4][16][384]
    __bf16* hL   = lds + TPW_ELEMS + A_ELEMS; // [WAVES*16][64]

    const int lane  = threadIdx.x & 31;
    const int wave  = threadIdx.x >> 5;
    const int mrow  = lane & 15;
    const int khalf = lane >> 4;
    const int nb    = (blockIdx.x * WAVES + wave) * 16;   // node base of this wave's tile

    // Warm the L2->WGP path for the packed weight stream (global_prefetch_b8).
    __builtin_prefetch(bpack + OFF_BG0 + (size_t)lane * 512, 0, 1);
    __builtin_prefetch(bpack + OFF_BG1 + (size_t)lane * 512, 0, 1);

    // ---------------- MLP stage (WMMA, bf16 x bf16 -> f32) ----------------
    // A1: inv feats 16x16 zero-padded to 16x32, K-layout per lane.
    v16bf A1;
    {
        const float* ip = invf + (size_t)(nb + mrow) * 16 + khalf * 8;
#pragma unroll
        for (int i = 0; i < 8; ++i) A1[i] = (__bf16)ip[i];
#pragma unroll
        for (int i = 8; i < 16; ++i) A1[i] = (__bf16)0.0f;
    }
    __bf16* hrow = hL + (size_t)(wave * 16 + mrow) * 64;

    // layer 1: 16x32 @ 32x64
#pragma unroll
    for (int j = 0; j < 4; ++j) {
        v16bf B = *(const v16bf*)(bpack + OFF_BM1 + (size_t)j * 512 + lane * 16);
        v8f d = {};
        d = WMMA_BF16(A1, B, d);
#pragma unroll
        for (int v = 0; v < 8; ++v) {
            int m = v + 8 * khalf;
            hL[(size_t)(wave * 16 + m) * 64 + j * 16 + (lane & 15)] = (__bf16)silu(d[v]);
        }
    }
    __syncthreads();

    // layers 2 and 3: 16x64 @ 64x64
#pragma unroll 1
    for (int layer = 0; layer < 2; ++layer) {
        const __bf16* BW = bpack + (layer ? OFF_BM3 : OFF_BM2);
        v16bf A[2];
#pragma unroll
        for (int ks = 0; ks < 2; ++ks) {
            int k0 = ks * 32 + khalf * 8;
            v8bf lo = *(const v8bf*)(hrow + k0);
            v8bf hi = *(const v8bf*)(hrow + k0 + 16);
            A[ks] = cat8(lo, hi);
        }
        __syncthreads();   // everyone done reading h before overwrite
#pragma unroll
        for (int j = 0; j < 4; ++j) {
            v8f d = {};
            v16bf B0 = *(const v16bf*)(BW + (size_t)(0 * 4 + j) * 512 + lane * 16);
            d = WMMA_BF16(A[0], B0, d);
            v16bf B1 = *(const v16bf*)(BW + (size_t)(1 * 4 + j) * 512 + lane * 16);
            d = WMMA_BF16(A[1], B1, d);
#pragma unroll
            for (int v = 0; v < 8; ++v) {
                int m = v + 8 * khalf;
                hL[(size_t)(wave * 16 + m) * 64 + j * 16 + (lane & 15)] = (__bf16)silu(d[v]);
            }
        }
        __syncthreads();
    }

    // layer 4: 16x64 @ 64x512 -> tpw (no activation), parked in LDS as bf16
    {
        v16bf A[2];
#pragma unroll
        for (int ks = 0; ks < 2; ++ks) {
            int k0 = ks * 32 + khalf * 8;
            v8bf lo = *(const v8bf*)(hrow + k0);
            v8bf hi = *(const v8bf*)(hrow + k0 + 16);
            A[ks] = cat8(lo, hi);
        }
#pragma unroll 1
        for (int j = 0; j < 32; ++j) {
            v8f d = {};
            v16bf B0 = *(const v16bf*)(bpack + OFF_BM4 + (size_t)(0 * 32 + j) * 512 + lane * 16);
            d = WMMA_BF16(A[0], B0, d);
            v16bf B1 = *(const v16bf*)(bpack + OFF_BM4 + (size_t)(1 * 32 + j) * 512 + lane * 16);
            d = WMMA_BF16(A[1], B1, d);
#pragma unroll
            for (int v = 0; v < 8; ++v) {
                int m = v + 8 * khalf;
                tpwL[(size_t)(wave * 16 + m) * 512 + j * 16 + (lane & 15)] = (__bf16)d[v];
            }
        }
    }
    __syncthreads();

    // ---------------- Elementwise stage: build four 16x384 A matrices ----------------
    // Two adjacent channels per lane -> packed bf16x2 DS stores (ds_store_b32).
    // 16 nodes x 64 channel-pairs = 1024 pairs per wave, 32 iterations of 32 lanes.
#pragma unroll 1
    for (int it = 0; it < 32; ++it) {
        int idx = it * 32 + lane;
        int m  = idx >> 6;          // node within tile (uniform per iteration)
        int c0 = (idx & 63) * 2;    // first channel of the pair
        int n = nb + m;

        // one-hot -> element id (uniform load stream)
        const float* at = node_attrs + (size_t)n * 10;
        int eid = 0;
#pragma unroll
        for (int e = 1; e < 10; ++e) if (at[e] > 0.5f) eid = e;
        float a0  = mga[(size_t)n * 4 + 0];
        float a1x = mga[(size_t)n * 4 + 1];
        float a1y = mga[(size_t)n * 4 + 2];
        float a1z = mga[(size_t)n * 4 + 3];

        const __bf16* tr = tpwL + (size_t)(wave * 16 + m) * 512;

        float mid0a[2], mid0b[2], y0v[2];
        float m1cx[2], m1cy[2], m1cz[2];
        float m1dx[2], m1dy[2], m1dz[2];
        float y1xv[2], y1yv[2], y1zv[2];
#pragma unroll
        for (int t = 0; t < 2; ++t) {
            int c = c0 + t;
            float4 nf = *(const float4*)(node_feats + ((size_t)n * 128 + c) * 4);
            float x0 = nf.x, x1 = nf.y, x2 = nf.z, x3 = nf.w;
            float nrm = x1 * x1 + x2 * x2 + x3 * x3;
            float x0sq = x0 * x0;

            const float* w0 = w_sc0 + (size_t)eid * 5 * C + c;
            float y0 = w0[0] * x0 + w0[C] * x0sq + w0[2 * C] * nrm + w0[3 * C] * (x0sq * x0) + w0[4 * C] * (x0 * nrm);
            const float* w1 = w_sc1 + (size_t)eid * 4 * C + c;
            float s1 = w1[0] + w1[C] * x0 + w1[2 * C] * x0sq + w1[3 * C] * nrm;
            float y1x = s1 * x1, y1y = s1 * x2, y1z = s1 * x3;

            float wa = (float)tr[c], wb = (float)tr[128 + c], wc = (float)tr[256 + c], wd = (float)tr[384 + c];

            float wcy = wc * y0;
            float wda = wd * a0;
            mid0a[t] = wa * y0 * a0;
            mid0b[t] = wb * (y1x * a1x + y1y * a1y + y1z * a1z);
            y0v[t] = y0;
            m1cx[t] = wcy * a1x;  m1cy[t] = wcy * a1y;  m1cz[t] = wcy * a1z;
            m1dx[t] = wda * y1x;  m1dy[t] = wda * y1y;  m1dz[t] = wda * y1z;
            y1xv[t] = y1x;        y1yv[t] = y1y;        y1zv[t] = y1z;
        }

        __bf16* r0 = aL + (size_t)((wave * 4 + 0) * 16 + m) * 384;
        __bf16* r1 = aL + (size_t)((wave * 4 + 1) * 16 + m) * 384;
        __bf16* r2 = aL + (size_t)((wave * 4 + 2) * 16 + m) * 384;
        __bf16* r3 = aL + (size_t)((wave * 4 + 3) * 16 + m) * 384;
        *(v2bf*)(r0 + c0)       = v2bf{(__bf16)mid0a[0], (__bf16)mid0a[1]};
        *(v2bf*)(r0 + 128 + c0) = v2bf{(__bf16)mid0b[0], (__bf16)mid0b[1]};
        *(v2bf*)(r0 + 256 + c0) = v2bf{(__bf16)y0v[0],   (__bf16)y0v[1]};
        *(v2bf*)(r1 + c0)       = v2bf{(__bf16)m1cx[0],  (__bf16)m1cx[1]};
        *(v2bf*)(r1 + 128 + c0) = v2bf{(__bf16)m1dx[0],  (__bf16)m1dx[1]};
        *(v2bf*)(r1 + 256 + c0) = v2bf{(__bf16)y1xv[0],  (__bf16)y1xv[1]};
        *(v2bf*)(r2 + c0)       = v2bf{(__bf16)m1cy[0],  (__bf16)m1cy[1]};
        *(v2bf*)(r2 + 128 + c0) = v2bf{(__bf16)m1dy[0],  (__bf16)m1dy[1]};
        *(v2bf*)(r2 + 256 + c0) = v2bf{(__bf16)y1yv[0],  (__bf16)y1yv[1]};
        *(v2bf*)(r3 + c0)       = v2bf{(__bf16)m1cz[0],  (__bf16)m1cz[1]};
        *(v2bf*)(r3 + 128 + c0) = v2bf{(__bf16)m1dz[0],  (__bf16)m1dz[1]};
        *(v2bf*)(r3 + 256 + c0) = v2bf{(__bf16)y1zv[0],  (__bf16)y1zv[1]};
    }
    __syncthreads();

    // ---------------- Main GEMMs: [mid | y] (16x384) @ [W_l;Wo] (384x128) ----------------
#pragma unroll 1
    for (int mm = 0; mm < 4; ++mm) {
        const __bf16* arow = aL + (size_t)((wave * 4 + mm) * 16 + mrow) * 384;
        v16bf Af[12];
#pragma unroll
        for (int ks = 0; ks < 12; ++ks) {
            int k0 = ks * 32 + khalf * 8;
            v8bf lo = *(const v8bf*)(arow + k0);
            v8bf hi = *(const v8bf*)(arow + k0 + 16);
            Af[ks] = cat8(lo, hi);
        }
        const __bf16* BG = bpack + (mm == 0 ? OFF_BG0 : OFF_BG1);
#pragma unroll 1
        for (int j = 0; j < 8; ++j) {
            v8f d = {};
#pragma unroll
            for (int ks = 0; ks < 12; ++ks) {
                v16bf B = *(const v16bf*)(BG + (size_t)(ks * 8 + j) * 512 + lane * 16);
                d = WMMA_BF16(Af[ks], B, d);
            }
#pragma unroll
            for (int v = 0; v < 8; ++v) {
                int m = v + 8 * khalf;
                int n = nb + m;
                int ccol = j * 16 + (lane & 15);
                int outcol = (mm == 0) ? ccol : (128 + ccol * 3 + (mm - 1));
                out[(size_t)n * 512 + outcol] = d[v] + sc[(size_t)n * 512 + outcol];
            }
        }
    }
}

extern "C" void kernel_launch(void* const* d_in, const int* in_sizes, int n_in,
                              void* d_out, int out_size, void* d_ws, size_t ws_size,
                              hipStream_t stream) {
    (void)in_sizes; (void)n_in; (void)out_size; (void)ws_size;
    const float* node_feats = (const float*)d_in[0];
    const float* sc_p       = (const float*)d_in[1];
    const float* node_attrs = (const float*)d_in[2];
    const float* invf       = (const float*)d_in[3];
    const float* mga        = (const float*)d_in[4];
    const float* w_sc0      = (const float*)d_in[5];
    const float* w_sc1      = (const float*)d_in[6];
    const float* w_mlp1     = (const float*)d_in[7];
    const float* w_mlp2     = (const float*)d_in[8];
    const float* w_mlp3     = (const float*)d_in[9];
    const float* w_mlp4     = (const float*)d_in[10];
    const float* W_l0       = (const float*)d_in[11];
    const float* W_l1       = (const float*)d_in[12];
    const float* Wo0        = (const float*)d_in[13];
    const float* Wo1        = (const float*)d_in[14];
    float* out = (float*)d_out;
    __bf16* bpack = (__bf16*)d_ws;

    // Pack weight matrices into WMMA B-fragment layout (bf16), L2-resident.
    pack_b_kernel<<<1 * 4, 32, 0, stream>>>(w_mlp1, 16, nullptr, 0, 4, bpack + OFF_BM1);
    pack_b_kernel<<<2 * 4, 32, 0, stream>>>(w_mlp2, 64, nullptr, 0, 4, bpack + OFF_BM2);
    pack_b_kernel<<<2 * 4, 32, 0, stream>>>(w_mlp3, 64, nullptr, 0, 4, bpack + OFF_BM3);
    pack_b_kernel<<<2 * 32, 32, 0, stream>>>(w_mlp4, 64, nullptr, 0, 32, bpack + OFF_BM4);
    pack_b_kernel<<<12 * 8, 32, 0, stream>>>(W_l0, 256, Wo0, 128, 8, bpack + OFF_BG0);
    pack_b_kernel<<<12 * 8, 32, 0, stream>>>(W_l1, 256, Wo1, 128, 8, bpack + OFF_BG1);

    // Fused MLP + elementwise + equivariant GEMM kernel.
    const int grid = N_NODES / (16 * WAVES);   // 1024 workgroups
    fused_kernel<<<grid, 32 * WAVES, LDS_BYTES, stream>>>(
        node_feats, sc_p, node_attrs, invf, mga, w_sc0, w_sc1, bpack, out);
}
